// JointRetriveDeformHead_30674656428639
// MI455X (gfx1250) — matrix operand order repaired and checked by reading, not compiled, not run on hardware
//
#include <hip/hip_runtime.h>

// ---------------------------------------------------------------------------
// CDNA5 / gfx1250 WMMA helpers (wave32). D = A(16x32 bf16) x B(32x16 bf16) + C(f32)
// ---------------------------------------------------------------------------
typedef __attribute__((ext_vector_type(16))) __bf16 v16bf;
typedef __attribute__((ext_vector_type(8)))  float  v8f;

__device__ __forceinline__ v8f wmma_bf16f32(v16bf a, v16bf b, v8f c) {
  // args: (neg_a, A, neg_b, B, c_mod, C, reuse_a, reuse_b)
  return __builtin_amdgcn_wmma_f32_16x16x32_bf16(false, a, false, b, (short)0, c,
                                                 false, false);
}

__device__ __forceinline__ v8f zero_v8f() {
  v8f z = {0.f, 0.f, 0.f, 0.f, 0.f, 0.f, 0.f, 0.f};
  return z;
}

// A fragment (16x32 bf16) from row-major tile, leading dim ldk.
// Layout: M = lane&15 ; K(e) = (e%8) + 16*(e/8) + 8*(lane>>4)
__device__ __forceinline__ v16bf load_a_frag(const __bf16* base, int ldk, int lane) {
  const __bf16* rp = base + (lane & 15) * ldk + ((lane >> 4) << 3);
  v16bf a;
#pragma unroll
  for (int e = 0; e < 8; ++e) a[e] = rp[e];
#pragma unroll
  for (int e = 0; e < 8; ++e) a[8 + e] = rp[16 + e];
  return a;
}

// B fragment from PRE-SWIZZLED weight storage: each lane's 16 bf16 are
// contiguous (32B), tiles ordered [kt][nt][lane][e]. One b128x2 load.
__device__ __forceinline__ v16bf load_b_frag_sw(const __bf16* sw, int ntiles,
                                                int kt, int nt, int lane) {
  return *(const v16bf*)(sw + ((((size_t)kt * ntiles + nt) << 5) + lane) * 16);
}

// ---------------------------------------------------------------------------
// Kernel 0: swizzle f32 weights [Kr x Nr] into bf16 WMMA-B fragment order,
// zero-padded to [Kp x Np].  out[((kt*ntiles+nt)*32+lane)*16+e] = W[k][n],
// k = kt*32 + (e%8) + 16*(e/8) + 8*(lane>>4), n = nt*16 + (lane&15).
// ---------------------------------------------------------------------------
__global__ void __launch_bounds__(256)
swizzle_weight(const float* __restrict__ W, __bf16* __restrict__ out,
               int Kr, int Kp, int Nr, int Np)
{
  int gid = blockIdx.x * 256 + threadIdx.x;
  if (gid >= Kp * Np) return;
  int e    = gid & 15;
  int lane = (gid >> 4) & 31;
  int tile = gid >> 9;
  int ntiles = Np >> 4;
  int nt = tile % ntiles;
  int kt = tile / ntiles;
  int k = kt * 32 + (e & 7) + ((e >> 3) << 4) + ((lane >> 4) << 3);
  int n = nt * 16 + (lane & 15);
  float v = (k < Kr && n < Nr) ? W[k * Nr + n] : 0.f;
  out[gid] = (__bf16)v;
}

// ---------------------------------------------------------------------------
// Kernel 1: fused PointNet encoder (per block: one batch x one encoder).
// 3->64->128->256 relu MLP, max-pool over 1024 points. All activations in LDS.
// Weights arrive pre-swizzled bf16. Dynamic LDS layout (bytes):
// ---------------------------------------------------------------------------
#define PN_OFF_W1 0          // swizzled [32x64]   bf16 : 4096
#define PN_OFF_W2 4096       // swizzled [64x128]  bf16 : 16384
#define PN_OFF_W3 20480      // swizzled [128x256] bf16 : 65536
#define PN_OFF_B1 86016      // [64]  f32 : 256
#define PN_OFF_B2 86272      // [128] f32 : 512
#define PN_OFF_B3 86784      // [256] f32 : 1024
#define PN_OFF_A  87808      // [128][32]  bf16 : 8192
#define PN_OFF_H1 96000      // [128][64]  bf16 : 16384
#define PN_OFF_H2 112384     // [128][128] bf16 : 32768
#define PN_OFF_GM 145152     // [256] int (running max bits) : 1024
#define PN_SMEM   146176

__global__ void __launch_bounds__(256)
pointnet_fused(const float* __restrict__ noc,
               const __bf16* __restrict__ teW1s, const float* __restrict__ teB1,
               const __bf16* __restrict__ teW2s, const float* __restrict__ teB2,
               const __bf16* __restrict__ teW3s, const float* __restrict__ teB3,
               const __bf16* __restrict__ reW1s, const float* __restrict__ reB1,
               const __bf16* __restrict__ reW2s, const float* __restrict__ reB2,
               const __bf16* __restrict__ reW3s, const float* __restrict__ reB3,
               __bf16* __restrict__ g_out)  // [2][64][256] bf16
{
  extern __shared__ __align__(16) char smem[];
  __bf16* sW1 = (__bf16*)(smem + PN_OFF_W1);
  __bf16* sW2 = (__bf16*)(smem + PN_OFF_W2);
  __bf16* sW3 = (__bf16*)(smem + PN_OFF_W3);
  float*  sB1 = (float*)(smem + PN_OFF_B1);
  float*  sB2 = (float*)(smem + PN_OFF_B2);
  float*  sB3 = (float*)(smem + PN_OFF_B3);
  __bf16* sA  = (__bf16*)(smem + PN_OFF_A);
  __bf16* sH1 = (__bf16*)(smem + PN_OFF_H1);
  __bf16* sH2 = (__bf16*)(smem + PN_OFF_H2);
  int*    sGM = (int*)(smem + PN_OFF_GM);

  const int b    = blockIdx.x;
  const int enc  = blockIdx.y;
  const int tid  = threadIdx.x;
  const int lane = tid & 31;   // wave32 on gfx1250
  const int wave = tid >> 5;

  const __bf16* W1 = enc ? reW1s : teW1s;  const float* B1 = enc ? reB1 : teB1;
  const __bf16* W2 = enc ? reW2s : teW2s;  const float* B2 = enc ? reB2 : teB2;
  const __bf16* W3 = enc ? reW3s : teW3s;  const float* B3 = enc ? reB3 : teB3;

  // stage pre-swizzled weights: contiguous b128 copies
  {
    uint4* d1 = (uint4*)sW1; const uint4* s1 = (const uint4*)W1;
    for (int i = tid; i < 256; i += 256) d1[i] = s1[i];      // 4 KB
    uint4* d2 = (uint4*)sW2; const uint4* s2 = (const uint4*)W2;
    for (int i = tid; i < 1024; i += 256) d2[i] = s2[i];     // 16 KB
    uint4* d3 = (uint4*)sW3; const uint4* s3 = (const uint4*)W3;
    for (int i = tid; i < 4096; i += 256) d3[i] = s3[i];     // 64 KB
  }
  if (tid < 64)  sB1[tid] = B1[tid];
  if (tid < 128) sB2[tid] = B2[tid];
  sB3[tid] = B3[tid];
  sGM[tid] = 0;  // post-relu values are >= 0, so int-max on bits == float-max
  __syncthreads();

  const float* noc_b = noc + (size_t)b * 3 * 1024;  // [3][1024]

  for (int chunk = 0; chunk < 8; ++chunk) {
    const int p0 = chunk * 128;
    // stage pts chunk [128][32] (cols >= 3 zero): pts[p][c] = noc[b][c][p]
    for (int i = tid; i < 128 * 32; i += 256) {
      int r = i >> 5, c = i & 31;
      sA[i] = (__bf16)((c < 3) ? noc_b[c * 1024 + p0 + r] : 0.f);
    }
    __syncthreads();

    // Layer 1: [128x32] x [32x64] -> H1, relu
    for (int t = wave; t < 32; t += 8) {
      int mt = t & 7, nt = t >> 3;
      v16bf a  = load_a_frag(sA + mt * 16 * 32, 32, lane);
      v16bf bB = load_b_frag_sw(sW1, 4, 0, nt, lane);
      v8f acc = zero_v8f();
      acc = wmma_bf16f32(a, bB, acc);
      int n = nt * 16 + (lane & 15);
      float bb = sB1[n];
      __bf16* out = sH1 + (mt * 16 + ((lane >> 4) << 3)) * 64 + n;
#pragma unroll
      for (int r = 0; r < 8; ++r) out[r * 64] = (__bf16)fmaxf(acc[r] + bb, 0.f);
    }
    __syncthreads();

    // Layer 2: [128x64] x [64x128] -> H2, relu
    for (int t = wave; t < 64; t += 8) {
      int mt = t & 7, nt = t >> 3;
      v8f acc = zero_v8f();
#pragma unroll
      for (int kk = 0; kk < 2; ++kk) {
        v16bf a  = load_a_frag(sH1 + mt * 16 * 64 + kk * 32, 64, lane);
        v16bf bB = load_b_frag_sw(sW2, 8, kk, nt, lane);
        acc = wmma_bf16f32(a, bB, acc);
      }
      int n = nt * 16 + (lane & 15);
      float bb = sB2[n];
      __bf16* out = sH2 + (mt * 16 + ((lane >> 4) << 3)) * 128 + n;
#pragma unroll
      for (int r = 0; r < 8; ++r) out[r * 128] = (__bf16)fmaxf(acc[r] + bb, 0.f);
    }
    __syncthreads();

    // Layer 3: [128x128] x [128x256], relu, fold into running max
    for (int t = wave; t < 128; t += 8) {
      int mt = t & 7, nt = t >> 3;
      v8f acc = zero_v8f();
#pragma unroll
      for (int kk = 0; kk < 4; ++kk) {
        v16bf a  = load_a_frag(sH2 + mt * 16 * 128 + kk * 32, 128, lane);
        v16bf bB = load_b_frag_sw(sW3, 16, kk, nt, lane);
        acc = wmma_bf16f32(a, bB, acc);
      }
      int n = nt * 16 + (lane & 15);
      float bb = sB3[n];
      float lmax = 0.f;
#pragma unroll
      for (int r = 0; r < 8; ++r) lmax = fmaxf(lmax, fmaxf(acc[r] + bb, 0.f));
      atomicMax(&sGM[n], __float_as_int(lmax));
    }
    __syncthreads();
  }
  g_out[((size_t)enc * 64 + b) * 256 + tid] = (__bf16)__int_as_float(sGM[tid]);
}

// ---------------------------------------------------------------------------
// Kernel 2: generic WMMA GEMM. out = [relu](A_bf16[MxK] @ W + bias)
// W is pre-swizzled bf16 ([Kp x Np] fragment order, L2-resident; fragments are
// direct global b128 loads). A chunk staged in dynamic LDS (128*K*2 bytes).
// K % 32 == 0, Np % 16 == 0, rows multiple of 8.
// ---------------------------------------------------------------------------
__global__ void __launch_bounds__(256)
gemm_bf16(const __bf16* __restrict__ A, int M, int K,
          const __bf16* __restrict__ Wsw, const float* __restrict__ bias,
          float* __restrict__ outF, __bf16* __restrict__ outB,
          int Nr, int Np, int relu)
{
  extern __shared__ __align__(16) char smem_g[];
  __bf16* Asm = (__bf16*)smem_g;

  const int tid = threadIdx.x, lane = tid & 31, wave = tid >> 5;
  const int row0 = blockIdx.x * 128;
  int rows = M - row0; if (rows > 128) rows = 128;

  {
    const uint4* As = (const uint4*)(A + (size_t)row0 * K);
    uint4* Ad = (uint4*)Asm;
    const int n16 = rows * K / 8;  // 8 bf16 per uint4
    for (int i = tid; i < n16; i += 256) Ad[i] = As[i];
  }
  __syncthreads();

  const int mtiles = (rows + 15) >> 4;
  const int ntiles = Np >> 4;
  for (int t = wave; t < mtiles * ntiles; t += 8) {
    int mt = t % mtiles, nt = t / mtiles;
    int n = nt * 16 + (lane & 15);
    v8f acc = zero_v8f();
    for (int k0 = 0; k0 < K; k0 += 32) {
      v16bf a  = load_a_frag(Asm + mt * 16 * K + k0, K, lane);
      v16bf bB = load_b_frag_sw(Wsw, ntiles, k0 >> 5, nt, lane);
      acc = wmma_bf16f32(a, bB, acc);
    }
    float bb = (n < Nr) ? bias[n] : 0.f;
#pragma unroll
    for (int r = 0; r < 8; ++r) {
      int m = row0 + mt * 16 + r + ((lane >> 4) << 3);
      if (m < M && n < Nr) {
        float v = acc[r] + bb;
        if (relu) v = fmaxf(v, 0.f);
        if (outF) outF[(size_t)m * Nr + n] = v;
        else      outB[(size_t)m * Nr + n] = (__bf16)v;
      }
    }
  }
}

// ---------------------------------------------------------------------------
// Kernel 3: per-batch variance-weighted distances over 1024 sources + top-10.
// ---------------------------------------------------------------------------
__global__ void __launch_bounds__(256)
dist_topk(const float* __restrict__ feat,   // [128][256]; ret encoder rows at +64
          const float* __restrict__ rc,     // ret_src_codes [1024][256]
          const float* __restrict__ var,    // src_variances [1024][256]
          int* __restrict__ idxb)           // [64][10]
{
  __shared__ float retv[256];
  __shared__ float ds[1024];
  __shared__ float rv[256];
  __shared__ int   ri[256];
  const int b = blockIdx.x, tid = threadIdx.x;

  retv[tid] = feat[(size_t)(64 + b) * 256 + tid];
  __syncthreads();

  for (int s = tid; s < 1024; s += 256) {
    const float4* rcp = (const float4*)(rc + (size_t)s * 256);
    const float4* vp  = (const float4*)(var + (size_t)s * 256);
    float acc = 0.f;
#pragma unroll 4
    for (int d4 = 0; d4 < 64; ++d4) {
      float4 r4 = rcp[d4];
      float4 v4 = vp[d4];
      float d0 = retv[4 * d4 + 0] - r4.x;
      float d1 = retv[4 * d4 + 1] - r4.y;
      float d2 = retv[4 * d4 + 2] - r4.z;
      float d3 = retv[4 * d4 + 3] - r4.w;
      acc += v4.x * d0 * d0 + v4.y * d1 * d1 + v4.z * d2 * d2 + v4.w * d3 * d3;
    }
    ds[s] = acc;
  }
  __syncthreads();

  for (int k = 0; k < 10; ++k) {
    float bv = 3.4e38f; int bi = 1 << 30;
    for (int s = tid; s < 1024; s += 256)
      if (ds[s] < bv || (ds[s] == bv && s < bi)) { bv = ds[s]; bi = s; }
    rv[tid] = bv; ri[tid] = bi;
    __syncthreads();
    for (int off = 128; off > 0; off >>= 1) {
      if (tid < off) {
        if (rv[tid + off] < rv[tid] ||
            (rv[tid + off] == rv[tid] && ri[tid + off] < ri[tid])) {
          rv[tid] = rv[tid + off]; ri[tid] = ri[tid + off];
        }
      }
      __syncthreads();
    }
    if (tid == 0) { idxb[b * 10 + k] = ri[0]; ds[ri[0]] = 3.4e38f; }
    __syncthreads();
  }
}

// ---------------------------------------------------------------------------
// Kernel 4: build decoder input x[10240][544] bf16 = [tgt(256)|src(256)|part(32)]
// rows ordered (b, k, part). One row per block.
// ---------------------------------------------------------------------------
__global__ void __launch_bounds__(256)
build_decoder_input(const float* __restrict__ feat,        // tgt rows 0..63
                    const float* __restrict__ src_codes,   // [1024][256]
                    const float* __restrict__ part_latent, // [1024][16][32]
                    const int* __restrict__ idxb,
                    __bf16* __restrict__ x)
{
  const int row = blockIdx.x;
  const int tid = threadIdx.x;
  const int b    = row / 160;
  const int rem  = row - b * 160;
  const int k    = rem >> 4;
  const int part = rem & 15;
  const int s = idxb[b * 10 + k];
  const float* tg = feat + (size_t)b * 256;
  const float* sc = src_codes + (size_t)s * 256;
  const float* pl = part_latent + ((size_t)s * 16 + part) * 32;
  __bf16* xr = x + (size_t)row * 544;
  for (int col = tid; col < 544; col += 256) {
    float v = (col < 256) ? tg[col]
            : (col < 512) ? sc[col - 256]
                          : pl[col - 512];
    xr[col] = (__bf16)v;
  }
}

// ---------------------------------------------------------------------------
// Kernel 5: per-(b,k): params = proj[s] @ raw + def[s]; out = mat[s] @ params.
// The mat gather (~377 MB) is the HBM-bound tail; float4 loads -> b128 stream.
// ---------------------------------------------------------------------------
__global__ void __launch_bounds__(256)
deform_apply(const float* __restrict__ rawbuf,  // [640][96] (10240 rows x 6)
             const int* __restrict__ idxb,      // [640]
             const float* __restrict__ proj,    // [1024][96][96]
             const float* __restrict__ defp,    // [1024][96]
             const float* __restrict__ mat,     // [1024][1536][96]
             float* __restrict__ out)           // [640][1536] == [64,10,512,3]
{
  __shared__ float raw_s[96];
  __shared__ float par_s[96];
  const int bk = blockIdx.x, tid = threadIdx.x;
  const int s = idxb[bk];

  if (tid < 96) raw_s[tid] = rawbuf[(size_t)bk * 96 + tid];
  __syncthreads();
  if (tid < 96) {
    float acc = defp[(size_t)s * 96 + tid];
    const float4* pr = (const float4*)(proj + ((size_t)s * 96 + tid) * 96);
#pragma unroll 4
    for (int j4 = 0; j4 < 24; ++j4) {
      float4 p4 = pr[j4];
      acc += p4.x * raw_s[4 * j4 + 0] + p4.y * raw_s[4 * j4 + 1] +
             p4.z * raw_s[4 * j4 + 2] + p4.w * raw_s[4 * j4 + 3];
    }
    par_s[tid] = acc;
  }
  __syncthreads();
  for (int p = tid; p < 1536; p += 256) {
    const float4* mr = (const float4*)(mat + ((size_t)s * 1536 + p) * 96);
    float acc = 0.f;
#pragma unroll 6
    for (int j4 = 0; j4 < 24; ++j4) {
      float4 m4 = mr[j4];
      acc += m4.x * par_s[4 * j4 + 0] + m4.y * par_s[4 * j4 + 1] +
             m4.z * par_s[4 * j4 + 2] + m4.w * par_s[4 * j4 + 3];
    }
    out[(size_t)bk * 1536 + p] = acc;
  }
}

// ---------------------------------------------------------------------------
// Host launcher. Workspace layout (bytes, ~22.9 MB total).
// ---------------------------------------------------------------------------
#define WS_G    0u           // g_bf   [128][256] bf16 : 65536
#define WS_FEAT 65536u       // feat   [128][256] f32  : 131072
#define WS_IDX  196608u      // idx    [640] int       : 2560
#define WS_RAW  199168u      // raw    [10240*6] f32   : 245760
#define WS_X    444928u      // xdec   [10240][544] bf16 : 11141120
#define WS_H1   11586048u    // h1     [10240][256] bf16 : 5242880
#define WS_H2   16828928u    // h2     [10240][256] bf16 : 5242880
#define WS_SW   22071808u    // swizzled bf16 weights (see launcher)

extern "C" void kernel_launch(void* const* d_in, const int* in_sizes, int n_in,
                              void* d_out, int out_size, void* d_ws, size_t ws_size,
                              hipStream_t stream) {
  const float* noc  = (const float*)d_in[0];
  const float* teW1 = (const float*)d_in[1];  const float* teB1 = (const float*)d_in[2];
  const float* teW2 = (const float*)d_in[3];  const float* teB2 = (const float*)d_in[4];
  const float* teW3 = (const float*)d_in[5];  const float* teB3 = (const float*)d_in[6];
  const float* teWf = (const float*)d_in[7];  const float* teBf = (const float*)d_in[8];
  const float* reW1 = (const float*)d_in[9];  const float* reB1 = (const float*)d_in[10];
  const float* reW2 = (const float*)d_in[11]; const float* reB2 = (const float*)d_in[12];
  const float* reW3 = (const float*)d_in[13]; const float* reB3 = (const float*)d_in[14];
  const float* reWf = (const float*)d_in[15]; const float* reBf = (const float*)d_in[16];
  const float* dW1  = (const float*)d_in[17]; const float* dB1  = (const float*)d_in[18];
  const float* dW2  = (const float*)d_in[19]; const float* dB2  = (const float*)d_in[20];
  const float* dW3  = (const float*)d_in[21]; const float* dB3  = (const float*)d_in[22];
  const float* ret_src   = (const float*)d_in[23];
  const float* src_codes = (const float*)d_in[24];
  const float* variances = (const float*)d_in[25];
  const float* part_lat  = (const float*)d_in[26];
  const float* def_param = (const float*)d_in[27];
  const float* proj      = (const float*)d_in[28];
  const float* mat       = (const float*)d_in[29];

  char* ws = (char*)d_ws;
  __bf16* g_bf  = (__bf16*)(ws + WS_G);
  float*  feat  = (float*)(ws + WS_FEAT);
  int*    idxb  = (int*)(ws + WS_IDX);
  float*  rawb  = (float*)(ws + WS_RAW);
  __bf16* xdec  = (__bf16*)(ws + WS_X);
  __bf16* h1    = (__bf16*)(ws + WS_H1);
  __bf16* h2    = (__bf16*)(ws + WS_H2);

  // swizzled-weight arena
  char* p = ws + WS_SW;
  auto alloc_sw = [&](size_t elems) { __bf16* r = (__bf16*)p; p += elems * 2; return r; };
  __bf16* teW1s = alloc_sw(32 * 64);
  __bf16* teW2s = alloc_sw(64 * 128);
  __bf16* teW3s = alloc_sw(128 * 256);
  __bf16* teWfs = alloc_sw(256 * 256);
  __bf16* reW1s = alloc_sw(32 * 64);
  __bf16* reW2s = alloc_sw(64 * 128);
  __bf16* reW3s = alloc_sw(128 * 256);
  __bf16* reWfs = alloc_sw(256 * 256);
  __bf16* dW1s  = alloc_sw(544 * 256);
  __bf16* dW2s  = alloc_sw(256 * 256);
  __bf16* dW3s  = alloc_sw(256 * 16);

  // 0) pre-swizzle all weight matrices into WMMA B-fragment order (bf16)
  auto swz = [&](const float* W, __bf16* o, int Kr, int Kp, int Nr, int Np) {
    int total = Kp * Np;
    swizzle_weight<<<(total + 255) / 256, 256, 0, stream>>>(W, o, Kr, Kp, Nr, Np);
  };
  swz(teW1, teW1s, 3, 32, 64, 64);     swz(reW1, reW1s, 3, 32, 64, 64);
  swz(teW2, teW2s, 64, 64, 128, 128);  swz(reW2, reW2s, 64, 64, 128, 128);
  swz(teW3, teW3s, 128, 128, 256, 256); swz(reW3, reW3s, 128, 128, 256, 256);
  swz(teWf, teWfs, 256, 256, 256, 256); swz(reWf, reWfs, 256, 256, 256, 256);
  swz(dW1, dW1s, 544, 544, 256, 256);
  swz(dW2, dW2s, 256, 256, 256, 256);
  swz(dW3, dW3s, 256, 256, 6, 16);

  // 1) fused PointNet encoders (grid: 64 batches x 2 encoders)
  pointnet_fused<<<dim3(64, 2), 256, PN_SMEM, stream>>>(
      noc, teW1s, teB1, teW2s, teB2, teW3s, teB3,
      reW1s, reB1, reW2s, reB2, reW3s, reB3, g_bf);

  // 2) final FC per encoder: [64x256] @ [256x256] -> feat f32
  gemm_bf16<<<1, 256, 128 * 256 * 2, stream>>>(
      g_bf, 64, 256, teWfs, teBf, feat, nullptr, 256, 256, 0);
  gemm_bf16<<<1, 256, 128 * 256 * 2, stream>>>(
      g_bf + 64 * 256, 64, 256, reWfs, reBf, feat + 64 * 256, nullptr, 256, 256, 0);

  // 3) distances + top-10 retrieval
  dist_topk<<<64, 256, 0, stream>>>(feat, ret_src, variances, idxb);

  // 4) decoder input gather -> bf16 [10240][544]
  build_decoder_input<<<10240, 256, 0, stream>>>(feat, src_codes, part_lat, idxb,
                                                 xdec);

  // 5) decoder MLP: 544 -> 256 (relu) -> 256 (relu) -> 6
  gemm_bf16<<<80, 256, 128 * 544 * 2, stream>>>(
      xdec, 10240, 544, dW1s, dB1, nullptr, h1, 256, 256, 1);
  gemm_bf16<<<80, 256, 128 * 256 * 2, stream>>>(
      h1, 10240, 256, dW2s, dB2, nullptr, h2, 256, 256, 1);
  gemm_bf16<<<80, 256, 128 * 256 * 2, stream>>>(
      h2, 10240, 256, dW3s, dB3, rawb, nullptr, 6, 16, 0);

  // 6) constraint projection + deformation matvec -> [64,10,512,3]
  deform_apply<<<640, 256, 0, stream>>>(rawb, idxb, proj, def_param, mat,
                                        (float*)d_out);
}
